// RNN_16260746183329
// MI455X (gfx1250) — compile-verified, matrix operand order checked by reading, not run on hardware
//
#include <hip/hip_runtime.h>
#include <hip/hip_bf16.h>

// ---------------------------------------------------------------------------
// Types for CDNA5 WMMA (wave32): 16x16x32 bf16 -> f32 accumulate
// ---------------------------------------------------------------------------
typedef __attribute__((ext_vector_type(16))) __bf16  v16bf;
typedef __attribute__((ext_vector_type(8)))  float   v8f;
typedef __attribute__((ext_vector_type(4)))  unsigned int u32x4;
typedef __attribute__((ext_vector_type(4)))  int      v4i;

typedef __attribute__((address_space(1))) v4i gas_v4i;   // global int4
typedef __attribute__((address_space(3))) v4i las_v4i;   // LDS int4

#if defined(__HIP_DEVICE_COMPILE__) && __has_builtin(__builtin_amdgcn_global_load_async_to_lds_b128)
#define HAVE_ASYNC_LDS 1
#else
#define HAVE_ASYNC_LDS 0
#endif

#if HAVE_ASYNC_LDS
template <int N>
__device__ __forceinline__ void wait_async() {
#if __has_builtin(__builtin_amdgcn_s_wait_asynccnt)
  __builtin_amdgcn_s_wait_asynccnt(N);
#else
  if (N == 0) asm volatile("s_wait_asynccnt 0x0" ::: "memory");
  else        asm volatile("s_wait_asynccnt 0x4" ::: "memory");
#endif
}

__device__ __forceinline__ void async_copy16(const __hip_bfloat16* gsrc,
                                             __hip_bfloat16* ldst) {
  void* gv = (void*)gsrc;   // drop const; builtin param is non-const
  void* lv = (void*)ldst;
  __builtin_amdgcn_global_load_async_to_lds_b128((gas_v4i*)gv, (las_v4i*)lv, 0, 0);
}
#endif

// Fragment loader for a row-major [rows x ld] bf16 matrix (global or LDS).
// CDNA5 16-bit A-matrix 16x32 lane layout (ISA 7.12.2):
//   lanes 0-15  : row M = lane,    K = {k0..k0+7, k0+16..k0+23}
//   lanes 16-31 : row M = lane-16, K = {k0+8..k0+15, k0+24..k0+31}
// B fragment (K x 16, column-per-lane) uses the same pattern on the
// pre-transposed [N][K] weight matrix.
__device__ __forceinline__ v16bf load_frag(const __hip_bfloat16* base, int ld,
                                           int row, int k0, int lane) {
  const __hip_bfloat16* p =
      base + (size_t)(row + (lane & 15)) * ld + k0 + ((lane >> 4) << 3);
  union { u32x4 u[2]; v16bf v; } t;
  t.u[0] = *(const u32x4*)(p);        // 8 bf16 = 16 bytes
  t.u[1] = *(const u32x4*)(p + 16);   // next 8 bf16
  return t.v;
}

__device__ __forceinline__ float sigf(float x) {
  return 1.0f / (1.0f + __expf(-x));
}

// ---------------------------------------------------------------------------
// Elementwise f32 -> bf16
// ---------------------------------------------------------------------------
__global__ void cvt_f32_bf16_kernel(const float* __restrict__ in,
                                    __hip_bfloat16* __restrict__ out,
                                    size_t n) {
  size_t i = (size_t)blockIdx.x * blockDim.x + threadIdx.x;
  if (i < n) out[i] = __float2bfloat16(in[i]);
}

// f32 [slices][R][C] -> bf16 [slices][C][R]  (per-slice transpose)
__global__ void cvt_transpose_kernel(const float* __restrict__ in,
                                     __hip_bfloat16* __restrict__ out,
                                     int slices, int R, int C) {
  size_t total = (size_t)slices * R * C;
  size_t i = (size_t)blockIdx.x * blockDim.x + threadIdx.x;
  if (i >= total) return;
  size_t rc = (size_t)R * C;
  int s = (int)(i / rc);
  size_t rem = i - (size_t)s * rc;
  int r = (int)(rem / C);
  int c = (int)(rem % C);
  out[(size_t)s * rc + (size_t)c * R + r] = __float2bfloat16(in[i]);
}

// ---------------------------------------------------------------------------
// Tile staging: copy one 128x32 bf16 tile of A and of B (row stride K in
// global, 32 in LDS).  256 threads, each moves 2x16B per tile.  On CDNA5 this
// uses GLOBAL_LOAD_ASYNC_TO_LDS_B128 (ASYNCcnt-tracked, no VGPR round-trip).
// ---------------------------------------------------------------------------
__device__ __forceinline__ void stage_tiles(const __hip_bfloat16* __restrict__ Ag,
                                            const __hip_bfloat16* __restrict__ Bg,
                                            __hip_bfloat16* sA, __hip_bfloat16* sB,
                                            int K, int tid) {
#pragma unroll
  for (int i = 0; i < 2; ++i) {
    const int c   = tid + (i << 8);     // 0..511 chunk id
    const int row = c >> 2;             // 0..127
    const int off = (c & 3) * 8;        // element offset (8 bf16 = 16 B)
    const __hip_bfloat16* ga = Ag + (size_t)row * K + off;
    const __hip_bfloat16* gb = Bg + (size_t)row * K + off;
    __hip_bfloat16* la = sA + row * 32 + off;
    __hip_bfloat16* lb = sB + row * 32 + off;
#if HAVE_ASYNC_LDS
    async_copy16(ga, la);
    async_copy16(gb, lb);
#else
    *(u32x4*)la = *(const u32x4*)ga;
    *(u32x4*)lb = *(const u32x4*)gb;
#endif
  }
}

// ---------------------------------------------------------------------------
// Input-projection GEMM:  C[M,N] = A[M,K] * Bt[N,K]^T + bias[N]   (bf16 in,
// f32 WMMA accumulate, bf16 out).  Block tile 128x128, 8 waves of 64x32.
// A/B tiles double-buffered in LDS via async copy; fragments via ds_load_b128.
// ---------------------------------------------------------------------------
__global__ __launch_bounds__(256) void gemm_bias_kernel(
    const __hip_bfloat16* __restrict__ A,
    const __hip_bfloat16* __restrict__ Bt,
    const float* __restrict__ bias,
    __hip_bfloat16* __restrict__ C,
    int M, int N, int K) {
  __shared__ __hip_bfloat16 sA[2][128 * 32];   // 8 KB each
  __shared__ __hip_bfloat16 sB[2][128 * 32];   // 8 KB each -> 32 KB total

  const int tid  = threadIdx.x;
  const int lane = tid & 31;
  const int wave = tid >> 5;                   // 0..7
  const int wy   = wave >> 2;                  // 2 waves along M (64 rows each)
  const int wx   = wave & 3;                   // 4 waves along N (32 cols each)
  const int m0   = blockIdx.x * 128 + wy * 64;
  const int n0   = blockIdx.y * 128 + wx * 32;

  const __hip_bfloat16* Ag = A  + (size_t)(blockIdx.x * 128) * K;
  const __hip_bfloat16* Bg = Bt + (size_t)(blockIdx.y * 128) * K;

  v8f zero = {};
  v8f acc[4][2];
#pragma unroll
  for (int i = 0; i < 4; ++i)
#pragma unroll
    for (int j = 0; j < 2; ++j) acc[i][j] = zero;

  const int nsteps = K / 32;
  stage_tiles(Ag, Bg, sA[0], sB[0], K, tid);   // prefetch k-step 0

  for (int kt = 0; kt < nsteps; ++kt) {
    const int cur = kt & 1;
    if (kt + 1 < nsteps)
      stage_tiles(Ag + (kt + 1) * 32, Bg + (kt + 1) * 32,
                  sA[cur ^ 1], sB[cur ^ 1], K, tid);
#if HAVE_ASYNC_LDS
    // 4 newer (next-buffer) loads may stay in flight; older 4 (current buffer)
    // must be complete.  Async loads retire in order.
    if (kt + 1 < nsteps) wait_async<4>(); else wait_async<0>();
#endif
    __syncthreads();

    v16bf a[4], b[2];
#pragma unroll
    for (int i = 0; i < 4; ++i)
      a[i] = load_frag(sA[cur], 32, wy * 64 + i * 16, 0, lane);
#pragma unroll
    for (int j = 0; j < 2; ++j)
      b[j] = load_frag(sB[cur], 32, wx * 32 + j * 16, 0, lane);
#pragma unroll
    for (int i = 0; i < 4; ++i)
#pragma unroll
      for (int j = 0; j < 2; ++j)
        acc[i][j] = __builtin_amdgcn_wmma_f32_16x16x32_bf16(
            false, a[i], false, b[j], (short)0, acc[i][j], false, false);

    __syncthreads();   // all waves done reading before buffer is rewritten
  }

  // C/D layout: VGPR r holds M = r (lanes 0-15) or r+8 (lanes 16-31), N = lane&15
  const int col  = lane & 15;
  const int rofs = (lane >> 4) << 3;
#pragma unroll
  for (int j = 0; j < 2; ++j) {
    const int n  = n0 + j * 16 + col;
    const float bv = bias[n];
#pragma unroll
    for (int i = 0; i < 4; ++i)
#pragma unroll
      for (int r = 0; r < 8; ++r) {
        const int m = m0 + i * 16 + rofs + r;
        C[(size_t)m * N + n] = __float2bfloat16(acc[i][j][r] + bv);
      }
  }
}

// ---------------------------------------------------------------------------
// Persistent recurrent kernel: one block per direction (grid = 2).
// 1024 threads = 32 waves; wave w owns N-tiles {4w..4w+3} of the 16x2048
// per-step GEMM (M = B = 16 = one WMMA tile).  LDS: h (16x512 bf16) and
// z (16x2048 f32) -> 144 KB, fits the 320 KB WGP LDS (impossible on CDNA4).
// Gate order i,f,g,o; all activations sigmoid (per reference).
// ---------------------------------------------------------------------------
__global__ __launch_bounds__(1024) void lstm_recurrent_kernel(
    const __hip_bfloat16* __restrict__ xz_all,  // [2][B*T][2048] bf16
    const __hip_bfloat16* __restrict__ Uht,     // [2][2048][512] bf16 (transposed)
    __hip_bfloat16* __restrict__ y_out,         // [B][T][1024] bf16 (concat fwd|bwd)
    float* __restrict__ out_f32,                // nullable: [B][T][1024] f32
    int T) {
  constexpr int Bb = 16, Uu = 512, Gg = 2048;
  extern __shared__ char smem[];
  __hip_bfloat16* hsm = (__hip_bfloat16*)smem;            // 16*512 bf16 = 16 KB
  float* zbuf = (float*)(smem + (size_t)Bb * Uu * 2);     // 16*2048 f32 = 128 KB

  const int dir  = blockIdx.x;
  const int tid  = threadIdx.x;
  const int lane = tid & 31;
  const int wave = tid >> 5;                               // 0..31

  const __hip_bfloat16* xz = xz_all + (size_t)dir * Bb * T * Gg;
  const __hip_bfloat16* Ub = Uht + (size_t)dir * Gg * Uu;

  // h0 = 0
  for (int i = tid; i < Bb * Uu; i += 1024) hsm[i] = __float2bfloat16(0.0f);
  // c0 = 0 (cell state lives in registers, fixed (row,unit) ownership per thread)
  float c[8];
#pragma unroll
  for (int j = 0; j < 8; ++j) c[j] = 0.0f;
  __syncthreads();

  const int col  = lane & 15;
  const int rofs = (lane >> 4) << 3;
  v8f zero = {};

  for (int s = 0; s < T; ++s) {
    const int t = dir ? (T - 1 - s) : s;    // scan is time-ordered either way

    // ---- phase 1: z = xz[t] + h @ Uh  (WMMA over K=512) ----
    v8f acc[4];
#pragma unroll
    for (int j = 0; j < 4; ++j) acc[j] = zero;

    for (int k0 = 0; k0 < Uu; k0 += 32) {
      v16bf a = load_frag(hsm, Uu, 0, k0, lane);          // ds_load_b128 x2
#pragma unroll
      for (int j = 0; j < 4; ++j) {
        const int n0 = (wave * 4 + j) * 16;
        v16bf b = load_frag(Ub, Uu, n0, k0, lane);        // L2-resident weights
        acc[j] = __builtin_amdgcn_wmma_f32_16x16x32_bf16(
            false, a, false, b, (short)0, acc[j], false, false);
      }
    }
#pragma unroll
    for (int j = 0; j < 4; ++j) {
      const int n = (wave * 4 + j) * 16 + col;
#pragma unroll
      for (int r = 0; r < 8; ++r) {
        const int brow = rofs + r;
        const float xv = __bfloat162float(xz[((size_t)brow * T + t) * Gg + n]);
        zbuf[brow * Gg + n] = acc[j][r] + xv;
      }
    }
    __syncthreads();

    // ---- phase 2: gates + state update; 8 (row,unit) pairs per thread ----
#pragma unroll
    for (int j = 0; j < 8; ++j) {
      const int flat = tid * 8 + j;
      const int u    = flat & (Uu - 1);
      const int brow = flat >> 9;
      const float* zr = zbuf + (size_t)brow * Gg;
      const float i_ = sigf(zr[u]);
      const float f_ = sigf(zr[Uu + u]);
      const float g_ = sigf(zr[2 * Uu + u]);   // activation='sigmoid'
      const float o_ = sigf(zr[3 * Uu + u]);
      c[j] = f_ * c[j] + i_ * g_;
      const float h = o_ * sigf(c[j]);         // output activation sigmoid
      hsm[brow * Uu + u] = __float2bfloat16(h);
      const size_t oidx = ((size_t)brow * T + t) * (2 * Uu) + (size_t)dir * Uu + u;
      y_out[oidx] = __float2bfloat16(h);
      if (out_f32) out_f32[oidx] = h;
    }
    __syncthreads();   // protect zbuf/hsm before next step's phase 1
  }
}

// ---------------------------------------------------------------------------
// Host-side orchestration
// ---------------------------------------------------------------------------
extern "C" void kernel_launch(void* const* d_in, const int* in_sizes, int n_in,
                              void* d_out, int out_size, void* d_ws, size_t ws_size,
                              hipStream_t stream) {
  (void)in_sizes; (void)n_in; (void)out_size; (void)ws_size;
  const float* x  = (const float*)d_in[0];   // [16,1024,1024]
  const float* W  = (const float*)d_in[1];   // [5,2,1024,2048]
  const float* Uh = (const float*)d_in[2];   // [5,2,512,2048]
  const float* bb = (const float*)d_in[3];   // [5,2,2048]

  const int B = 16, T = 1024, D = 1024, L = 5, U = 512, G = 4 * U;  // G = 2048
  const int M = B * T;  // 16384

  // workspace carve-out (~264 MB)
  char* ws = (char*)d_ws;
  size_t off = 0;
  auto carve = [&](size_t bytes) -> char* {
    char* p = ws + off;
    off += (bytes + 255) & ~(size_t)255;
    return p;
  };
  __hip_bfloat16* Wt  = (__hip_bfloat16*)carve((size_t)L * 2 * G * D * 2); // [l,d][N=2048][K=1024]
  __hip_bfloat16* Uht = (__hip_bfloat16*)carve((size_t)L * 2 * G * U * 2); // [l,d][N=2048][K=512]
  __hip_bfloat16* yA  = (__hip_bfloat16*)carve((size_t)B * T * D * 2);
  __hip_bfloat16* yB  = (__hip_bfloat16*)carve((size_t)B * T * D * 2);
  __hip_bfloat16* xz  = (__hip_bfloat16*)carve((size_t)2 * B * T * G * 2); // [dir][B*T][2048]

  // one-time conversions
  {
    size_t n = (size_t)B * T * D;
    cvt_f32_bf16_kernel<<<(unsigned)((n + 255) / 256), 256, 0, stream>>>(x, yA, n);
  }
  {
    size_t n = (size_t)L * 2 * D * G;
    cvt_transpose_kernel<<<(unsigned)((n + 255) / 256), 256, 0, stream>>>(W, Wt, L * 2, D, G);
  }
  {
    size_t n = (size_t)L * 2 * U * G;
    cvt_transpose_kernel<<<(unsigned)((n + 255) / 256), 256, 0, stream>>>(Uh, Uht, L * 2, U, G);
  }

  const size_t smem = (size_t)B * U * 2 /*h bf16*/ + (size_t)B * G * 4 /*z f32*/; // 147456 B

  __hip_bfloat16* cur = yA;
  __hip_bfloat16* nxt = yB;
  for (int l = 0; l < L; ++l) {
    // input projections for both directions: big compute-bound GEMMs
    for (int d = 0; d < 2; ++d) {
      gemm_bias_kernel<<<dim3(M / 128, G / 128), 256, 0, stream>>>(
          cur, Wt + (size_t)(l * 2 + d) * G * D, bb + (size_t)(l * 2 + d) * G,
          xz + (size_t)d * B * T * G, M, G, D);
    }
    // recurrence: both directions concurrently (one persistent block each)
    lstm_recurrent_kernel<<<2, 1024, (unsigned)smem, stream>>>(
        xz, Uht + (size_t)(l * 2) * G * U, nxt,
        (l == L - 1) ? (float*)d_out : nullptr, T);
    __hip_bfloat16* tmp = cur; cur = nxt; nxt = tmp;
  }
}
